// Linear2Bit_17102559773071
// MI455X (gfx1250) — matrix operand order repaired and checked by reading, not compiled, not run on hardware
//
#include <hip/hip_runtime.h>
#include <hip/hip_bf16.h>

// ---------------------------------------------------------------------------
// 2-bit quantized linear:  out[m,n] = sum_k x[m,k] * w[n,k] + bias[n]
//   M = 256 (8*32), N = 16384, K = 4096, group size 16.
//   w[n,k] = norm[g] * (2q-3)/3, q = 2-bit code, g = n*256 + k/16.
//
// MI455X plan (weight-BW-bound skinny GEMM, ~80MB @ 23.3TB/s ≈ 3.5us):
//   * Pre-pass converts x fp32->f16 once into d_ws.
//   * Each WG: all 256 rows x 64 output cols -> weights stream exactly once.
//   * A tile (256x32 f16 = 16KB) staged in LDS, double buffered, via
//     GLOBAL_LOAD_ASYNC_TO_LDS_B128 (ASYNCcnt) issued one K-step ahead;
//     80B row pitch -> conflict-free ds_load_b128 fragment reads.
//   * Dequant: 2KB LDS LUT (byte -> packed f16 multipliers in {-1,-1/3,1/3,1}),
//     ds_load_b64 + 2x v_pk_mul_f16 per int32, under the WMMA shadow.
//   * 8 waves = 4(M) x 2(N); 8x v_wmma_f32_16x16x32_f16 per K-step of 32.
// ---------------------------------------------------------------------------

#define AS1 __attribute__((address_space(1)))
#define AS3 __attribute__((address_space(3)))

typedef __attribute__((ext_vector_type(16))) _Float16     v16h;
typedef __attribute__((ext_vector_type(8)))  _Float16     v8h;
typedef __attribute__((ext_vector_type(2)))  _Float16     v2h;
typedef __attribute__((ext_vector_type(8)))  float        v8f;
typedef __attribute__((ext_vector_type(4)))  float        v4f;
typedef __attribute__((ext_vector_type(4)))  int          v4i;
typedef __attribute__((ext_vector_type(8)))  unsigned int v8u;

union FragU  { v8u u; v16h h; };
union PairU  { unsigned int u; v2h h; };
union FragHH { struct { v8h lo, hi; } p; v16h h; };

#define M_TOT 256
#define N_TOT 16384
#define K_TOT 4096
#define GRP_PER_ROW 256
#define ROW_B 80                       // padded LDS pitch: 20 dwords -> 20r mod 64
                                       // tiles all 64 banks for 16-lane b128 reads
#define ABUF_B (256 * ROW_B)           // 20KB per buffer

#if __has_builtin(__builtin_amdgcn_global_load_async_to_lds_b128)
#define HAVE_ASYNC_LDS 1
#else
#define HAVE_ASYNC_LDS 0
#endif

__device__ __forceinline__ void wait_async0() {
#if __has_builtin(__builtin_amdgcn_s_wait_asynccnt)
    __builtin_amdgcn_s_wait_asynccnt(0);
#else
    asm volatile("s_wait_asynccnt 0x0" ::: "memory");
#endif
}

// ---- Pre-pass: x fp32 -> f16 ------------------------------------------------
__global__ __launch_bounds__(256)
void Cvt_x_f16_kernel(const float* __restrict__ x, _Float16* __restrict__ xh) {
    const int i = (blockIdx.x * 256 + threadIdx.x) * 8;
    const v4f a = *(const v4f*)(x + i);
    const v4f b = *(const v4f*)(x + i + 4);
    v8h o;
#pragma unroll
    for (int j = 0; j < 4; ++j) { o[j] = (_Float16)a[j]; o[4 + j] = (_Float16)b[j]; }
    *(v8h*)(xh + i) = o;
}

// ---- Main GEMM --------------------------------------------------------------
__global__ __launch_bounds__(256)
void Linear2Bit_wmma_kernel(const _Float16* __restrict__ xh,     // [256,4096] f16
                            const int*      __restrict__ wq,     // [G,4]
                            const _Float16* __restrict__ wnorm,  // [G]
                            const float*    __restrict__ bias,   // [16384]
                            float*          __restrict__ out)    // [256,16384]
{
    __shared__ unsigned int lut[256][2];
    __shared__ __align__(16) char abuf[2][ABUF_B];

    {   // Dequant LUT: byte -> two packed-f16 multiplier pairs, m(q) = (2q-3)/3.
        const unsigned int b = threadIdx.x;
        const unsigned short mt[4] = {0xBC00u, 0xB555u, 0x3555u, 0x3C00u};
        const unsigned int q0 = b & 3, q1 = (b >> 2) & 3, q2 = (b >> 4) & 3, q3 = (b >> 6) & 3;
        lut[b][0] = (unsigned int)mt[q0] | ((unsigned int)mt[q1] << 16);
        lut[b][1] = (unsigned int)mt[q2] | ((unsigned int)mt[q3] << 16);
    }
    __syncthreads();

    const int tid    = threadIdx.x;
    const int lane   = tid & 31;
    const int wave   = tid >> 5;
    const int wave_m = wave & 3;
    const int wave_n = wave >> 2;
    const int lane_lo = lane & 15;
    const int lane_hi = lane >> 4;

    const int nbase = blockIdx.x * 64 + wave_n * 32;
    const int mbase = wave_m * 64;

    v8f acc[4][2];
#pragma unroll
    for (int mi = 0; mi < 4; ++mi)
#pragma unroll
        for (int ni = 0; ni < 2; ++ni) acc[mi][ni] = (v8f){};

    const long ocol[2] = {
        (long)(nbase + lane_lo)      * GRP_PER_ROW + lane_hi,
        (long)(nbase + 16 + lane_lo) * GRP_PER_ROW + lane_hi };

    // Cooperative A staging: thread t copies rows (t>>2)+64j, 16B chunk t&3.
    const int crow  = tid >> 2;                       // 0..63
    const int cpart = tid & 3;                        // 0..3
    const _Float16* gsrc = xh + (long)crow * K_TOT + cpart * 8;
    const int ldst = crow * ROW_B + cpart * 16;

#if !HAVE_ASYNC_LDS
    v4i sreg[4];
#pragma unroll
    for (int j = 0; j < 4; ++j)
        sreg[j] = *(const v4i*)(gsrc + (long)j * 64 * K_TOT);   // kb = 0
#else
#pragma unroll
    for (int j = 0; j < 4; ++j)
        __builtin_amdgcn_global_load_async_to_lds_b128(
            (AS1 v4i*)(gsrc + (long)j * 64 * K_TOT),
            (AS3 v4i*)(&abuf[0][ldst + j * 64 * ROW_B]), 0, 0);
#endif

    for (int kb = 0; kb < K_TOT; kb += 32) {
        const int cur = (kb >> 5) & 1;
        const int gcol = kb >> 4;

#if HAVE_ASYNC_LDS
        wait_async0();                 // my writes to abuf[cur] are in LDS
        __syncthreads();               // everyone's writes landed, reads retired
        if (kb + 32 < K_TOT) {         // stage next K-chunk one step ahead
#pragma unroll
            for (int j = 0; j < 4; ++j)
                __builtin_amdgcn_global_load_async_to_lds_b128(
                    (AS1 v4i*)(gsrc + (long)j * 64 * K_TOT + kb + 32),
                    (AS3 v4i*)(&abuf[cur ^ 1][ldst + j * 64 * ROW_B]), 0, 0);
        }
#else
#pragma unroll
        for (int j = 0; j < 4; ++j)
            *(v4i*)(&abuf[cur][ldst + j * 64 * ROW_B]) = sreg[j];
        asm volatile("s_wait_dscnt 0x0" ::: "memory");
        __syncthreads();
        if (kb + 32 < K_TOT) {
#pragma unroll
            for (int j = 0; j < 4; ++j)
                sreg[j] = *(const v4i*)(gsrc + (long)j * 64 * K_TOT + kb + 32);
        }
#endif

        // ---- B fragments: LDS LUT + pk_mul_f16 dequant ----------------------
        v16h bfrag[2];
#pragma unroll
        for (int ni = 0; ni < 2; ++ni) {
            const long g = ocol[ni] + gcol;
            const v4i q4 = *(const v4i*)(wq + g * 4);
            const _Float16 nh = wnorm[g];
            const v2h nm2 = {nh, nh};
            FragU B;
#pragma unroll
            for (int k2 = 0; k2 < 4; ++k2) {
                const unsigned int w = (unsigned int)q4[k2];  // codes in [0,255]
                PairU a0, a1, r0, r1;
                a0.u = lut[w][0];
                a1.u = lut[w][1];
                r0.h = a0.h * nm2;
                r1.h = a1.h * nm2;
                B.u[2 * k2]     = r0.u;
                B.u[2 * k2 + 1] = r1.u;
            }
            bfrag[ni] = B.h;
        }

        // Weight-stream prefetch ~4 K-steps ahead.
        if (kb + 128 < K_TOT) {
            __builtin_prefetch((const void*)(wq + (ocol[0] + gcol + 8) * 4), 0, 1);
            __builtin_prefetch((const void*)(wq + (ocol[1] + gcol + 8) * 4), 0, 1);
        }

        // ---- A fragments from LDS (conflict-free ds_load_b128) --------------
        v16h afrag[4];
#pragma unroll
        for (int mi = 0; mi < 4; ++mi) {
            const char* rp = &abuf[cur][(mbase + mi * 16 + lane_lo) * ROW_B + lane_hi * 32];
            FragHH A;
            A.p.lo = *(const v8h*)(rp);        // K kb+sel*16 .. +7
            A.p.hi = *(const v8h*)(rp + 16);   // K kb+sel*16+8 .. +15
            afrag[mi] = A.h;
        }

        // ---- 8 WMMAs --------------------------------------------------------
#pragma unroll
        for (int mi = 0; mi < 4; ++mi)
#pragma unroll
            for (int ni = 0; ni < 2; ++ni)
                acc[mi][ni] = __builtin_amdgcn_wmma_f32_16x16x32_f16(
                    false, afrag[mi], false, bfrag[ni],
                    (short)0, acc[mi][ni], false, false);
    }

    // ---- Epilogue: bias add + store ----------------------------------------
#pragma unroll
    for (int ni = 0; ni < 2; ++ni) {
        const int n = nbase + ni * 16 + lane_lo;
        const float bv = bias[n];
#pragma unroll
        for (int mi = 0; mi < 4; ++mi) {
            const int m0 = mbase + mi * 16 + lane_hi * 8;
#pragma unroll
            for (int r = 0; r < 8; ++r)
                out[(long)(m0 + r) * N_TOT + n] = acc[mi][ni][r] + bv;
        }
    }
}

extern "C" void kernel_launch(void* const* d_in, const int* in_sizes, int n_in,
                              void* d_out, int out_size, void* d_ws, size_t ws_size,
                              hipStream_t stream) {
    const float*    x     = (const float*)d_in[0];
    const int*      wq    = (const int*)d_in[1];
    const _Float16* wnorm = (const _Float16*)d_in[2];
    const float*    bias  = (const float*)d_in[3];
    float*          out   = (float*)d_out;
    _Float16*       xh    = (_Float16*)d_ws;    // 2MB scratch

    Cvt_x_f16_kernel<<<512, 256, 0, stream>>>(x, xh);
    Linear2Bit_wmma_kernel<<<N_TOT / 64, 256, 0, stream>>>(xh, wq, wnorm, bias, out);
}
